// PyTorchConv1dAutograd_86526411145859
// MI455X (gfx1250) — matrix operand order repaired and checked by reading, not compiled, not run on hardware
//
#include <hip/hip_runtime.h>

// CDNA5 / gfx1250: wave32, WMMA 16x16x32 bf16 -> f32.
typedef __attribute__((ext_vector_type(16))) __bf16 v16bf;
typedef __attribute__((ext_vector_type(8)))  float  v8f;

constexpr int N_ = 32, C_ = 8, W_ = 65536, F_ = 16, KW = 64;
constexpr int OUTW   = W_ - KW + 1;                    // 65473
constexpr int TSPAN  = 512;                            // output positions per block
constexpr int XSPAN  = TSPAN + KW;                     // staged x per channel (576)
constexpr int BLOCKS_PER_N = (OUTW + TSPAN - 1) / TSPAN; // 128
constexpr int THREADS = 256;                           // 8 wave32 waves

struct __align__(16) Smem {
  unsigned short wfrag[2][C_][2][32][16]; // [hi/lo][c][kblk][lane][elem] = 32 KB
  unsigned       xpk[C_][XSPAN];          // (bf16_hi << 16) | bf16_lo    = 18 KB
  float          bias[F_];
};

__device__ __forceinline__ v16bf load_afrag(const unsigned short* p) {
  union { v16bf v; uint4 q[2]; } u;
  const uint4* s = (const uint4*)p;     // 32B-aligned lane slots -> 2x ds_load_b128
  u.q[0] = s[0]; u.q[1] = s[1];
  return u.v;
}

__global__ __launch_bounds__(THREADS)
void conv1d_wmma_bf16x2(const float* __restrict__ x, const float* __restrict__ w,
                        const float* __restrict__ bias, float* __restrict__ out) {
  __shared__ Smem sm;
  const int tid = threadIdx.x;
  const int bid = blockIdx.x;
  const int n   = bid / BLOCKS_PER_N;
  const int T0  = (bid - n * BLOCKS_PER_N) * TSPAN;

  // ---- stage x span: split f32 -> (bf16 hi, bf16 lo) packed in one dword ----
  for (int i = tid; i < C_ * XSPAN; i += THREADS) {
    int c = i / XSPAN, idx = i - c * XSPAN;
    int g = T0 + idx;
    float v = (g < W_) ? x[((size_t)n * C_ + c) * W_ + g] : 0.0f;
    unsigned bits = __float_as_uint(v);
    unsigned hi   = bits & 0xFFFF0000u;                 // truncated bf16 (hi plane)
    float    rem  = v - __uint_as_float(hi);
    unsigned lo   = __float_as_uint(rem) >> 16;         // residual bf16 (lo plane)
    sm.xpk[c][idx] = hi | lo;
  }

  // ---- stage w pre-swizzled into per-lane A-fragment layout (ISA 7.12.2) ----
  // A 16x32 bf16: lane L (m=L&15, h=L>>4), element e -> K = 16h+e (code's own k order, not ISA table)
  for (int s = tid; s < 2 * C_ * 2 * 32; s += THREADS) {
    int lane = s & 31;
    int b    = (s >> 5) & 1;
    int c    = (s >> 6) & 7;
    int p    = (s >> 9) & 1;          // 0 = hi part, 1 = lo part
    int f = lane & 15, h = lane >> 4;
    #pragma unroll
    for (int e = 0; e < 16; ++e) {
      int K = 32 * b + 16 * h + e;
      float wv = w[((size_t)f * C_ + c) * KW + K];
      unsigned bits = __float_as_uint(wv);
      unsigned hi   = bits & 0xFFFF0000u;
      float    rem  = wv - __uint_as_float(hi);
      unsigned sel  = p ? (__float_as_uint(rem) & 0xFFFF0000u) : hi;
      sm.wfrag[p][c][b][lane][e] = (unsigned short)(sel >> 16);
    }
  }
  if (tid < F_) sm.bias[tid] = bias[tid];
  __syncthreads();

  const int lane = tid & 31;
  const int wv   = tid >> 5;        // wave id: 4 tiles of 16 outputs each
  const int h    = lane >> 4;
  const int nlo  = lane & 15;

  v8f acc[4];
  #pragma unroll
  for (int j = 0; j < 4; ++j)
    #pragma unroll
    for (int v = 0; v < 8; ++v) acc[j][v] = 0.0f;

  float bval[8];
  #pragma unroll
  for (int v = 0; v < 8; ++v) bval[v] = sm.bias[v + 8 * h];  // f = v + 8h

  for (int c = 0; c < C_; ++c) {
    // register-cache the 4 A fragments for this channel (reused by 4 tiles)
    v16bf Ah0 = load_afrag(&sm.wfrag[0][c][0][lane][0]);
    v16bf Al0 = load_afrag(&sm.wfrag[1][c][0][lane][0]);
    v16bf Ah1 = load_afrag(&sm.wfrag[0][c][1][lane][0]);
    v16bf Al1 = load_afrag(&sm.wfrag[1][c][1][lane][0]);
    #pragma unroll
    for (int j = 0; j < 4; ++j) {
      // B 32x16 bf16: lane L (n=L&15, h=L>>4), element e -> K = 16h + e
      // B[k][n] = x[t0 + n + 32b + k]
      const int tb = wv * 64 + j * 16 + nlo + 16 * h;
      #pragma unroll
      for (int b = 0; b < 2; ++b) {
        unsigned q[16];
        const unsigned* xr = &sm.xpk[c][tb + 32 * b];
        #pragma unroll
        for (int e = 0; e < 16; ++e) q[e] = xr[e];     // dword-aligned, read2-friendly
        union { v16bf v; unsigned u[8]; } bh, bl;
        #pragma unroll
        for (int k = 0; k < 8; ++k) {                  // v_perm extracts hi/lo planes
          bh.u[k] = __builtin_amdgcn_perm(q[2 * k + 1], q[2 * k], 0x07060302u);
          bl.u[k] = __builtin_amdgcn_perm(q[2 * k + 1], q[2 * k], 0x05040100u);
        }
        v16bf Ah = b ? Ah1 : Ah0;
        v16bf Al = b ? Al1 : Al0;
        // split product: (Ah+Al)(Bh+Bl) ~= AhBh + AhBl + AlBh   (f32 accumulate)
        acc[j] = __builtin_amdgcn_wmma_f32_16x16x32_bf16(false, Ah, false, bh.v,
                                                         (short)0, acc[j], false, false);
        acc[j] = __builtin_amdgcn_wmma_f32_16x16x32_bf16(false, Ah, false, bl.v,
                                                         (short)0, acc[j], false, false);
        acc[j] = __builtin_amdgcn_wmma_f32_16x16x32_bf16(false, Al, false, bh.v,
                                                         (short)0, acc[j], false, false);
      }
    }
  }

  // ---- epilogue: reference adds bias C_ times; mask the ragged tail ----
  const size_t outbase = (size_t)n * F_ * (size_t)OUTW;
  #pragma unroll
  for (int j = 0; j < 4; ++j) {
    int t = T0 + wv * 64 + j * 16 + nlo;
    if (t < OUTW) {
      #pragma unroll
      for (int v = 0; v < 8; ++v) {
        int f = v + 8 * h;                 // C/D layout: VGPR v, lane>=16 -> M=v+8
        out[outbase + (size_t)f * OUTW + t] = acc[j][v] + (float)C_ * bval[v];
      }
    }
  }
}

extern "C" void kernel_launch(void* const* d_in, const int* in_sizes, int n_in,
                              void* d_out, int out_size, void* d_ws, size_t ws_size,
                              hipStream_t stream) {
  (void)in_sizes; (void)n_in; (void)out_size; (void)d_ws; (void)ws_size;
  const float* x = (const float*)d_in[0];
  const float* w = (const float*)d_in[1];
  const float* b = (const float*)d_in[2];
  float* out = (float*)d_out;
  dim3 grid(N_ * BLOCKS_PER_N);   // 4096 blocks
  dim3 block(THREADS);            // 8 wave32 waves
  conv1d_wmma_bf16x2<<<grid, block, 0, stream>>>(x, w, b, out);
}
